// FlexSummaryAttentionBlock_14061722927970
// MI455X (gfx1250) — matrix operand order repaired and checked by reading, not compiled, not run on hardware
//
#include <hip/hip_runtime.h>

#define E_   1024
#define H_   16
#define KVH_ 4
#define HD_  64
#define FFN_ 4096
#define B_   2
#define S_   128
#define R_   2048
#define KV_  (KVH_*HD_)   /* 256 */
#define LK_  (S_ + R_)    /* 2176 */
#define SCALE_ 0.125f     /* 1/sqrt(64) */

typedef __attribute__((ext_vector_type(16))) _Float16 v16h;
typedef __attribute__((ext_vector_type(8)))  float    v8f;

union AFrag { v16h v; unsigned int u[8]; };
union BFrag { v16h v; uint4 q[2]; _Float16 h[16]; };

// ---------------- WMMA fragment loaders (16x16x32 f16, contiguous-k order used here) -------
// A (16x32, MxK): lanes 0-15 -> M=lane, K in {0..15};
//                 lanes 16-31 -> M=lane-16, K in {16..31}.
__device__ inline v16h load_a_frag(const _Float16* rowPtr, int half) {
  const unsigned int* p = (const unsigned int*)rowPtr;  // pairs of f16
  AFrag a;
  int pb = half * 8;
#pragma unroll
  for (int j = 0; j < 4; ++j) a.u[j] = p[pb + j];           // K = kb+2j
#pragma unroll
  for (int j = 4; j < 8; ++j) a.u[j] = p[pb + j];           // K = kb+2j
  return a.v;
}
// B (32x16, KxN) sourced from W stored (N,K) row-major:
// lane<16 -> N=lane, K=0..15 contiguous; lane>=16 -> N=lane-16, K=16..31.
__device__ inline v16h load_b_frag(const _Float16* colPtr, int half) {
  const uint4* p = (const uint4*)(colPtr + half * 16);
  BFrag b;
  b.q[0] = p[0];
  b.q[1] = p[1];
  return b.v;
}
__device__ inline v8f wmma_f16(v16h a, v16h b, v8f c) {
  return __builtin_amdgcn_wmma_f32_16x16x32_f16(false, a, false, b, (short)0, c, false, false);
}

// ------ register-blocked GEMM: C = base + A(M,K) @ W(N,K)^T + bias --------
// Each wave computes a 32(M) x 64(N) strip: 2 A-frags x 4 B-frags -> 8 WMMA
// per 32-wide K step (24 B moved per lane per WMMA).
__global__ void gemm16(const _Float16* __restrict__ A, const _Float16* __restrict__ W,
                       const float* __restrict__ bias, const float* __restrict__ base,
                       float* __restrict__ C, int M, int N, int K) {
  int tn = blockIdx.x * 64, tm = blockIdx.y * 32;
  int lane = threadIdx.x & 31;
  int half = lane >> 4, l15 = lane & 15;
  v8f acc[2][4] = {{{}, {}, {}, {}}, {{}, {}, {}, {}}};
  const _Float16* arow0 = A + (size_t)(tm + l15) * K;
  const _Float16* arow1 = A + (size_t)(tm + 16 + l15) * K;
  const _Float16* brow[4];
#pragma unroll
  for (int t = 0; t < 4; ++t) brow[t] = W + (size_t)(tn + t * 16 + l15) * K;
  for (int k0 = 0; k0 < K; k0 += 32) {
    if (k0 + 256 < K) {  // gfx1250 global_prefetch_b8 one block ahead
      __builtin_prefetch((const void*)(arow0 + k0 + 256), 0, 1);
      __builtin_prefetch((const void*)(arow1 + k0 + 256), 0, 1);
    }
    v16h a0 = load_a_frag(arow0 + k0, half);
    v16h a1 = load_a_frag(arow1 + k0, half);
#pragma unroll
    for (int t = 0; t < 4; ++t) {
      v16h b = load_b_frag(brow[t] + k0, half);
      acc[0][t] = wmma_f16(a0, b, acc[0][t]);
      acc[1][t] = wmma_f16(a1, b, acc[1][t]);
    }
  }
#pragma unroll
  for (int t = 0; t < 4; ++t) {
    int n = tn + t * 16 + l15;
    float bv = bias ? bias[n] : 0.0f;
#pragma unroll
    for (int mt = 0; mt < 2; ++mt) {
#pragma unroll
      for (int i = 0; i < 8; ++i) {
        int m = tm + mt * 16 + i + half * 8;
        float r = acc[mt][t][i] + bv;
        if (base) r += base[(size_t)m * N + n];
        C[(size_t)m * N + n] = r;
      }
    }
  }
}

// ------ fused SwiGLU up: Out16 = silu(A@Wg^T) * (A@Wu^T) ------------------
// Each wave computes 16(M) x 64(N): 1 A-frag x (4 Wg + 4 Wu) -> 8 WMMA/step.
__global__ void swiglu16(const _Float16* __restrict__ A, const _Float16* __restrict__ Wg,
                         const _Float16* __restrict__ Wu, _Float16* __restrict__ Out,
                         int M, int N, int K) {
  int tn = blockIdx.x * 64, tm = blockIdx.y * 16;
  int lane = threadIdx.x & 31;
  int half = lane >> 4, l15 = lane & 15;
  v8f accg[4] = {{}, {}, {}, {}}, accu[4] = {{}, {}, {}, {}};
  const _Float16* arow = A + (size_t)(tm + l15) * K;
  const _Float16* grow[4];
  const _Float16* urow[4];
#pragma unroll
  for (int t = 0; t < 4; ++t) {
    grow[t] = Wg + (size_t)(tn + t * 16 + l15) * K;
    urow[t] = Wu + (size_t)(tn + t * 16 + l15) * K;
  }
  for (int k0 = 0; k0 < K; k0 += 32) {
    if (k0 + 256 < K) __builtin_prefetch((const void*)(arow + k0 + 256), 0, 1);
    v16h a = load_a_frag(arow + k0, half);
#pragma unroll
    for (int t = 0; t < 4; ++t) {
      accg[t] = wmma_f16(a, load_b_frag(grow[t] + k0, half), accg[t]);
      accu[t] = wmma_f16(a, load_b_frag(urow[t] + k0, half), accu[t]);
    }
  }
#pragma unroll
  for (int t = 0; t < 4; ++t) {
    int n = tn + t * 16 + l15;
#pragma unroll
    for (int i = 0; i < 8; ++i) {
      int m = tm + i + half * 8;
      float g = accg[t][i], u = accu[t][i];
      Out[(size_t)m * N + n] = (_Float16)((g / (1.0f + __expf(-g))) * u);
    }
  }
}

// ---------------- RMSNorm (fp32 in, f16 out) -------------------------------
__global__ void rmsnorm16(const float* __restrict__ x, const float* __restrict__ w,
                          _Float16* __restrict__ out, int ncols) {
  int row = blockIdx.x, tid = threadIdx.x;
  const float* xr = x + (size_t)row * ncols;
  float s = 0.0f;
  for (int i = tid; i < ncols; i += blockDim.x) { float v = xr[i]; s += v * v; }
  __shared__ float red[256];
  red[tid] = s; __syncthreads();
  for (int off = 128; off > 0; off >>= 1) {
    if (tid < off) red[tid] += red[tid + off];
    __syncthreads();
  }
  float inv = rsqrtf(red[0] / (float)ncols + 1e-6f);
  _Float16* orow = out + (size_t)row * ncols;
  for (int i = tid; i < ncols; i += blockDim.x) orow[i] = (_Float16)(xr[i] * inv * w[i]);
}

// ---------------- fp32 -> f16 convert --------------------------------------
__global__ void cvt16(const float* __restrict__ in, _Float16* __restrict__ out, size_t n) {
  size_t i = (size_t)blockIdx.x * blockDim.x + threadIdx.x;
  size_t st = (size_t)gridDim.x * blockDim.x;
  for (; i < n; i += st) out[i] = (_Float16)in[i];
}

// ---------------- RoPE 1D: layout (T, nh*HD), pos = row % S ----------------
__global__ void rope1d16(const float* __restrict__ in, _Float16* __restrict__ out,
                         int T, int nh) {
  size_t n = (size_t)T * nh * HD_;
  size_t i = (size_t)blockIdx.x * blockDim.x + threadIdx.x;
  size_t st = (size_t)gridDim.x * blockDim.x;
  for (; i < n; i += st) {
    int f = (int)(i % (size_t)(nh * HD_));
    int row = (int)(i / (size_t)(nh * HD_));
    int d = f & (HD_ - 1);
    int s = row % S_;
    int fi = d & 31;
    float inv = __powf(10000.0f, -(float)fi / 32.0f);
    float fr = (float)s * inv;
    float c = __cosf(fr), sn = __sinf(fr);
    size_t base = i - (size_t)d;
    float rh = (d < 32) ? -in[base + d + 32] : in[base + d - 32];
    out[i] = (_Float16)(in[i] * c + rh * sn);
  }
}

// ---------------- RoPE 2D (hierarchical bar/token) -------------------------
__global__ void rope2d16(const float* __restrict__ in, _Float16* __restrict__ out,
                         const int* __restrict__ bar, const int* __restrict__ tok,
                         int T, int nh) {
  size_t n = (size_t)T * nh * HD_;
  size_t i = (size_t)blockIdx.x * blockDim.x + threadIdx.x;
  size_t st = (size_t)gridDim.x * blockDim.x;
  for (; i < n; i += st) {
    int f = (int)(i % (size_t)(nh * HD_));
    int row = (int)(i / (size_t)(nh * HD_));
    int d = f & (HD_ - 1);
    int b = row / R_, r = row % R_;
    int m = d & 31;
    int pos, e;
    if (m < 16) { pos = bar[(size_t)b * R_ + r]; e = m; }
    else        { pos = tok[(size_t)b * R_ + r]; e = m - 16; }
    float inv = __powf(10000.0f, -(float)e / 16.0f);
    float fr = (float)pos * inv;
    float c = __cosf(fr), sn = __sinf(fr);
    size_t base = i - (size_t)d;
    float rh = (d < 32) ? -in[base + d + 32] : in[base + d - 32];
    out[i] = (_Float16)(in[i] * c + rh * sn);
  }
}

// ---------------- analytic masks ------------------------------------------
__device__ inline bool attn_mask(int stage, int q, int j) {
  if (j < S_) return (stage == 1) ? (j <= q) : (j <= (q >> 4));
  int jr = j - S_;
  return (stage == 1) ? ((jr >> 4) == q) : (jr <= q);
}

// ---------------- fused flash attention (wave per b,h,16-q tile) ----------
// Q: (B, Lq, H*HD) f16; Ks/Vs: (B, S, KV) f16; Kr/Vr: (B, R, KV) f16
// Out: (B, Lq, H*HD) f32
__global__ void attn16(const _Float16* __restrict__ Q,
                       const _Float16* __restrict__ Ks, const _Float16* __restrict__ Kr,
                       const _Float16* __restrict__ Vs, const _Float16* __restrict__ Vr,
                       float* __restrict__ Out, int Lq, int stage) {
  __shared__ _Float16 pbuf[16 * 32];
  int lane = threadIdx.x & 31;
  int half = lane >> 4, l15 = lane & 15;
  int qtiles = Lq >> 4;
  int bt = blockIdx.x;
  int qt = bt % qtiles;
  int h = (bt / qtiles) % H_;
  int b = bt / (qtiles * H_);
  int q0 = qt * 16;
  int kvh = h / (H_ / KVH_);

  const _Float16* qrow = Q + ((size_t)(b * Lq + q0 + l15)) * E_ + h * HD_;
  v16h qa0 = load_a_frag(qrow, half);
  v16h qa1 = load_a_frag(qrow + 32, half);

  v8f acc[4] = {{}, {}, {}, {}};
  float mrun[8], lrun[8];
  int qrm[8];
#pragma unroll
  for (int i = 0; i < 8; ++i) { mrun[i] = -1e30f; lrun[i] = 0.0f; qrm[i] = q0 + i + half * 8; }

  for (int kb = 0; kb < LK_ / 32; ++kb) {
    int keybase = kb * 32;
    const _Float16 *Kp, *Vp;
    if (keybase < S_) {
      Kp = Ks + ((size_t)(b * S_ + keybase)) * KV_ + kvh * HD_;
      Vp = Vs + ((size_t)(b * S_ + keybase)) * KV_ + kvh * HD_;
    } else {
      Kp = Kr + ((size_t)(b * R_ + keybase - S_)) * KV_ + kvh * HD_;
      Vp = Vr + ((size_t)(b * R_ + keybase - S_)) * KV_ + kvh * HD_;
    }
    // scores: two 16x16 tiles over HD=64 (two K-steps each)
    v8f c0 = {}, c1 = {};
    {
      const _Float16* kr0 = Kp + (size_t)l15 * KV_;
      c0 = wmma_f16(qa0, load_b_frag(kr0, half), c0);
      c0 = wmma_f16(qa1, load_b_frag(kr0 + 32, half), c0);
      const _Float16* kr1 = Kp + (size_t)(16 + l15) * KV_;
      c1 = wmma_f16(qa0, load_b_frag(kr1, half), c1);
      c1 = wmma_f16(qa1, load_b_frag(kr1 + 32, half), c1);
    }
    int j0 = keybase + l15, j1 = keybase + 16 + l15;
#pragma unroll
    for (int i = 0; i < 8; ++i) {
      float s0 = attn_mask(stage, qrm[i], j0) ? c0[i] * SCALE_ : -1e30f;
      float s1 = attn_mask(stage, qrm[i], j1) ? c1[i] * SCALE_ : -1e30f;
      float rv = fmaxf(s0, s1);
      for (int msk = 8; msk; msk >>= 1) rv = fmaxf(rv, __shfl_xor(rv, msk, 16));
      float nm = fmaxf(mrun[i], rv);
      float fac = __expf(mrun[i] - nm);
      float p0 = (s0 > -1e29f) ? __expf(s0 - nm) : 0.0f;
      float p1 = (s1 > -1e29f) ? __expf(s1 - nm) : 0.0f;
      float rs = p0 + p1;
      for (int msk = 8; msk; msk >>= 1) rs += __shfl_xor(rs, msk, 16);
      lrun[i] = lrun[i] * fac + rs;
      mrun[i] = nm;
#pragma unroll
      for (int dt = 0; dt < 4; ++dt) acc[dt][i] *= fac;
      int m = i + half * 8;
      pbuf[m * 32 + l15]      = (_Float16)p0;
      pbuf[m * 32 + 16 + l15] = (_Float16)p1;
    }
    __syncthreads();
    v16h pA = load_a_frag(pbuf + l15 * 32, half);  // P as A-fragment (16x32)
#pragma unroll
    for (int dt = 0; dt < 4; ++dt) {
      BFrag vb;
#pragma unroll
      for (int j = 0; j < 16; ++j)
        vb.h[j] = Vp[(size_t)(half * 16 + j) * KV_ + dt * 16 + l15];
      acc[dt] = wmma_f16(pA, vb.v, acc[dt]);
    }
    __syncthreads();
  }
#pragma unroll
  for (int i = 0; i < 8; ++i) {
    float l = fmaxf(lrun[i], 1e-20f);
    size_t ob = ((size_t)(b * Lq + qrm[i])) * E_ + h * HD_;
#pragma unroll
    for (int dt = 0; dt < 4; ++dt) Out[ob + dt * 16 + l15] = acc[dt][i] / l;
  }
}

// ============================ host orchestration ===========================
extern "C" void kernel_launch(void* const* d_in, const int* in_sizes, int n_in,
                              void* d_out, int out_size, void* d_ws, size_t ws_size,
                              hipStream_t stream) {
  (void)in_sizes; (void)n_in; (void)out_size; (void)ws_size;
  const float* sum_x = (const float*)d_in[0];
  const float* reg_x = (const float*)d_in[1];
  const int* bar_ids = (const int*)d_in[2];
  const int* tok_ids = (const int*)d_in[3];
  // d_in[4], d_in[5] (bool masks) reproduced analytically on device
  const float* Wsq = (const float*)d_in[6];
  const float* Wsk = (const float*)d_in[7];
  const float* Wsv = (const float*)d_in[8];
  const float* Wso = (const float*)d_in[9];
  const float* Wrq = (const float*)d_in[10];
  const float* Wrk = (const float*)d_in[11];
  const float* Wrv = (const float*)d_in[12];
  const float* Wro = (const float*)d_in[13];
  const float* Wk2 = (const float*)d_in[14];
  const float* Wv2 = (const float*)d_in[15];
  const float* Wsg = (const float*)d_in[16];
  const float* Wsu = (const float*)d_in[17];
  const float* Wsd = (const float*)d_in[18];
  const float* Wrg = (const float*)d_in[19];
  const float* Wru = (const float*)d_in[20];
  const float* Wrd = (const float*)d_in[21];
  const float* bsq = (const float*)d_in[22];
  const float* bsk = (const float*)d_in[23];
  const float* bsv = (const float*)d_in[24];
  const float* bso = (const float*)d_in[25];
  const float* brq = (const float*)d_in[26];
  const float* brk = (const float*)d_in[27];
  const float* brv = (const float*)d_in[28];
  const float* bro = (const float*)d_in[29];
  const float* bk2 = (const float*)d_in[30];
  const float* bv2 = (const float*)d_in[31];
  const float* n_sa = (const float*)d_in[32];
  const float* n_ra = (const float*)d_in[33];
  const float* n_sf = (const float*)d_in[34];
  const float* n_rf = (const float*)d_in[35];

  const int Ts = B_ * S_;   // 256
  const int Tr = B_ * R_;   // 4096

  char* wsp = (char*)d_ws;
  auto alloc  = [&](size_t bytes) -> void* { void* p = wsp; wsp += (bytes + 255) & ~(size_t)255; return p; };
  auto allocH = [&](size_t n) -> _Float16* { return (_Float16*)alloc(n * sizeof(_Float16)); };
  auto allocF = [&](size_t n) -> float*    { return (float*)alloc(n * sizeof(float)); };

  // f16 weights
  _Float16* wsq = allocH((size_t)E_ * E_);   _Float16* wsk = allocH((size_t)KV_ * E_);
  _Float16* wsv = allocH((size_t)KV_ * E_);  _Float16* wso = allocH((size_t)E_ * E_);
  _Float16* wrq = allocH((size_t)E_ * E_);   _Float16* wrk = allocH((size_t)KV_ * E_);
  _Float16* wrv = allocH((size_t)KV_ * E_);  _Float16* wro = allocH((size_t)E_ * E_);
  _Float16* wk2 = allocH((size_t)KV_ * E_);  _Float16* wv2 = allocH((size_t)KV_ * E_);
  _Float16* wsg = allocH((size_t)FFN_ * E_); _Float16* wsu = allocH((size_t)FFN_ * E_);
  _Float16* wsd = allocH((size_t)E_ * FFN_); _Float16* wrg = allocH((size_t)FFN_ * E_);
  _Float16* wru = allocH((size_t)FFN_ * E_); _Float16* wrd = allocH((size_t)E_ * FFN_);
  // f16 activations
  _Float16* sn16 = allocH((size_t)Ts * E_);  _Float16* rn16 = allocH((size_t)Tr * E_);
  _Float16* sq16 = allocH((size_t)Ts * E_);  _Float16* sk16 = allocH((size_t)Ts * KV_);
  _Float16* sv16 = allocH((size_t)Ts * KV_); _Float16* rq16 = allocH((size_t)Tr * E_);
  _Float16* rk16 = allocH((size_t)Tr * KV_); _Float16* rv16 = allocH((size_t)Tr * KV_);
  _Float16* sa16 = allocH((size_t)Ts * E_);  _Float16* ra16 = allocH((size_t)Tr * E_);
  _Float16* k216 = allocH((size_t)Ts * KV_); _Float16* v216 = allocH((size_t)Ts * KV_);
  _Float16* sfn16 = allocH((size_t)Ts * E_); _Float16* rfn16 = allocH((size_t)Tr * E_);
  _Float16* hs16 = allocH((size_t)Ts * FFN_); _Float16* hr16 = allocH((size_t)Tr * FFN_);
  // f32 scratch
  float* sqf = allocF((size_t)Ts * E_);   float* skf = allocF((size_t)Ts * KV_);
  float* svf = allocF((size_t)Ts * KV_);  float* rqf = allocF((size_t)Tr * E_);
  float* rkf = allocF((size_t)Tr * KV_);  float* rvf = allocF((size_t)Tr * KV_);
  float* saf = allocF((size_t)Ts * E_);   float* k2f = allocF((size_t)Ts * KV_);
  float* v2f = allocF((size_t)Ts * KV_);  float* raf = allocF((size_t)Tr * E_);

  float* out_sum = (float*)d_out;
  float* out_reg = out_sum + (size_t)Ts * E_;

  auto cvt = [&](const float* in, _Float16* out, size_t n) {
    int blocks = (int)((n + 255) / 256); if (blocks > 4096) blocks = 4096;
    cvt16<<<dim3(blocks), dim3(256), 0, stream>>>(in, out, n);
  };
  auto gemm = [&](const _Float16* A, const _Float16* W, const float* bias,
                  const float* base, float* C, int M, int N, int K) {
    gemm16<<<dim3(N / 64, M / 32), dim3(32), 0, stream>>>(A, W, bias, base, C, M, N, K);
  };

  // 1. weights -> f16
  cvt(Wsq, wsq, (size_t)E_ * E_);   cvt(Wsk, wsk, (size_t)KV_ * E_);
  cvt(Wsv, wsv, (size_t)KV_ * E_);  cvt(Wso, wso, (size_t)E_ * E_);
  cvt(Wrq, wrq, (size_t)E_ * E_);   cvt(Wrk, wrk, (size_t)KV_ * E_);
  cvt(Wrv, wrv, (size_t)KV_ * E_);  cvt(Wro, wro, (size_t)E_ * E_);
  cvt(Wk2, wk2, (size_t)KV_ * E_);  cvt(Wv2, wv2, (size_t)KV_ * E_);
  cvt(Wsg, wsg, (size_t)FFN_ * E_); cvt(Wsu, wsu, (size_t)FFN_ * E_);
  cvt(Wsd, wsd, (size_t)E_ * FFN_); cvt(Wrg, wrg, (size_t)FFN_ * E_);
  cvt(Wru, wru, (size_t)FFN_ * E_); cvt(Wrd, wrd, (size_t)E_ * FFN_);

  // 2. pre-attention RMSNorm
  rmsnorm16<<<dim3(Ts), dim3(256), 0, stream>>>(sum_x, n_sa, sn16, E_);
  rmsnorm16<<<dim3(Tr), dim3(256), 0, stream>>>(reg_x, n_ra, rn16, E_);

  // 3. q/k/v projections
  gemm(sn16, wsq, bsq, nullptr, sqf, Ts, E_, E_);
  gemm(sn16, wsk, bsk, nullptr, skf, Ts, KV_, E_);
  gemm(sn16, wsv, bsv, nullptr, svf, Ts, KV_, E_);
  gemm(rn16, wrq, brq, nullptr, rqf, Tr, E_, E_);
  gemm(rn16, wrk, brk, nullptr, rkf, Tr, KV_, E_);
  gemm(rn16, wrv, brv, nullptr, rvf, Tr, KV_, E_);

  // 4. RoPE + f16 packing
  rope1d16<<<dim3(1024), dim3(256), 0, stream>>>(sqf, sq16, Ts, H_);
  rope1d16<<<dim3(256),  dim3(256), 0, stream>>>(skf, sk16, Ts, KVH_);
  cvt(svf, sv16, (size_t)Ts * KV_);
  rope2d16<<<dim3(4096), dim3(256), 0, stream>>>(rqf, rq16, bar_ids, tok_ids, Tr, H_);
  rope2d16<<<dim3(1024), dim3(256), 0, stream>>>(rkf, rk16, bar_ids, tok_ids, Tr, KVH_);
  cvt(rvf, rv16, (size_t)Tr * KV_);

  // 5. stage-1 attention (summary queries over [summary|regular] keys)
  attn16<<<dim3(B_ * H_ * (S_ / 16)), dim3(32), 0, stream>>>(
      sq16, sk16, rk16, sv16, rv16, saf, S_, 1);
  cvt(saf, sa16, (size_t)Ts * E_);

  // 6. summary residual + k2/v2
  gemm(sa16, wso, bso, sum_x, out_sum, Ts, E_, E_);
  gemm(sa16, wk2, bk2, nullptr, k2f, Ts, KV_, E_);
  gemm(sa16, wv2, bv2, nullptr, v2f, Ts, KV_, E_);
  cvt(k2f, k216, (size_t)Ts * KV_);
  cvt(v2f, v216, (size_t)Ts * KV_);

  // 7. stage-2 attention (regular queries over [summary2|regular] keys)
  attn16<<<dim3(B_ * H_ * (R_ / 16)), dim3(32), 0, stream>>>(
      rq16, k216, rk16, v216, rv16, raf, R_, 2);
  cvt(raf, ra16, (size_t)Tr * E_);
  gemm(ra16, wro, bro, reg_x, out_reg, Tr, E_, E_);

  // 8. FFN: pre-norm, fused SwiGLU up, down-proj accumulated into d_out
  rmsnorm16<<<dim3(Ts), dim3(256), 0, stream>>>(out_sum, n_sf, sfn16, E_);
  rmsnorm16<<<dim3(Tr), dim3(256), 0, stream>>>(out_reg, n_rf, rfn16, E_);
  swiglu16<<<dim3(FFN_ / 64, Ts / 16), dim3(32), 0, stream>>>(sfn16, wsg, wsu, hs16, Ts, FFN_, E_);
  swiglu16<<<dim3(FFN_ / 64, Tr / 16), dim3(32), 0, stream>>>(rfn16, wrg, wru, hr16, Tr, FFN_, E_);
  gemm(hs16, wsd, nullptr, out_sum, out_sum, Ts, E_, FFN_);
  gemm(hr16, wrd, nullptr, out_reg, out_reg, Tr, E_, FFN_);
}
